// Attention_4380866642258
// MI455X (gfx1250) — 16-bit and 8-bit matrix operand order repaired and checked by reading, compiled, not run on hardware
//
#include <hip/hip_runtime.h>
#include <hip/hip_bf16.h>
#include <stdint.h>

typedef __attribute__((ext_vector_type(16))) _Float16 v16h;
typedef __attribute__((ext_vector_type(8)))  float    v8f;
typedef __attribute__((ext_vector_type(8)))  int      v8i;
typedef __attribute__((ext_vector_type(4)))  int      v4i;

#define NTOK 9232
#define SEQ  577
#define CDIM 768
#define C3   2304
#define NH   12
#define HD   64
#define NBH  192
#define NT   37
#define EPSQ 1.1920928955078125e-07f

#if __has_builtin(__builtin_amdgcn_global_load_async_to_lds_b128)
#define USE_ASYNC 1
#else
#define USE_ASYNC 0
#endif

__device__ __forceinline__ void wait_async(){
#if __has_builtin(__builtin_amdgcn_s_wait_asynccnt)
  __builtin_amdgcn_s_wait_asynccnt(0);
#else
  asm volatile("s_wait_asynccnt 0" ::: "memory");
#endif
}

__device__ __forceinline__ void async_b128(const void* gsrc, void* lds){
#if USE_ASYNC
  __builtin_amdgcn_global_load_async_to_lds_b128(
      (__attribute__((address_space(1))) v4i*)const_cast<void*>(gsrc),
      (__attribute__((address_space(3))) v4i*)lds, 0, 0);
#else
  *(int4*)lds = *(const int4*)gsrc;
#endif
}

__device__ __forceinline__ unsigned fenc(float f){
  unsigned u = __float_as_uint(f);
  return (u & 0x80000000u) ? ~u : (u | 0x80000000u);
}
__device__ __forceinline__ float fdec(unsigned u){
  unsigned b = (u & 0x80000000u) ? (u & 0x7FFFFFFFu) : ~u;
  return __uint_as_float(b);
}
__device__ __forceinline__ float clampf(float v, float lo, float hi){
  return fminf(fmaxf(v, lo), hi);
}
__device__ __forceinline__ void qparams(const unsigned* stats, int idx, float& s, float& zp){
  float mn = fminf(fdec(stats[2*idx+0]), 0.0f);
  float mx = fmaxf(fdec(stats[2*idx+1]), 0.0f);
  s  = fmaxf((mx - mn) * (1.0f/255.0f), EPSQ);
  zp = -128.0f - rintf(mn / s);
}

__global__ void k_init_stats(unsigned* stats){
  int i = threadIdx.x;
  if (i < 6){ stats[2*i] = 0xFFFFFFFFu; stats[2*i+1] = 0u; }
}

__global__ void k_quant_w(const float* w_qkv, const float* w_proj,
                          int8_t* wq_qkv, int* wsum_qkv, float* wsc_qkv, float* wzp_qkv,
                          int8_t* wq_proj, int* wsum_proj, float* wsc_proj, float* wzp_proj){
  __shared__ float smn[256], smx[256];
  __shared__ int ssum;
  int r = blockIdx.x, t = threadIdx.x;
  const float* w; int8_t* wq; int* wsum; float* wsc; float* wzp; int row;
  if (r < C3){ w = w_qkv + (size_t)r*CDIM; wq = wq_qkv + (size_t)r*CDIM;
               wsum = wsum_qkv; wsc = wsc_qkv; wzp = wzp_qkv; row = r; }
  else       { int rr = r - C3; w = w_proj + (size_t)rr*CDIM; wq = wq_proj + (size_t)rr*CDIM;
               wsum = wsum_proj; wsc = wsc_proj; wzp = wzp_proj; row = rr; }
  float v0 = w[t], v1 = w[t+256], v2 = w[t+512];
  smn[t] = fminf(fminf(v0, v1), fminf(v2, 0.0f));
  smx[t] = fmaxf(fmaxf(v0, v1), fmaxf(v2, 0.0f));
  if (t == 0) ssum = 0;
  __syncthreads();
  for (int off = 128; off; off >>= 1){
    if (t < off){ smn[t] = fminf(smn[t], smn[t+off]); smx[t] = fmaxf(smx[t], smx[t+off]); }
    __syncthreads();
  }
  float s  = fmaxf((smx[0]-smn[0]) * (1.0f/255.0f), EPSQ);
  float zp = -128.0f - rintf(smn[0] / s);
  int q0 = (int)clampf(rintf(v0/s)+zp, -128.f, 127.f);
  int q1 = (int)clampf(rintf(v1/s)+zp, -128.f, 127.f);
  int q2 = (int)clampf(rintf(v2/s)+zp, -128.f, 127.f);
  wq[t] = (int8_t)q0; wq[t+256] = (int8_t)q1; wq[t+512] = (int8_t)q2;
  atomicAdd(&ssum, q0+q1+q2);
  __syncthreads();
  if (t == 0){ wsum[row] = ssum; wsc[row] = s; wzp[row] = zp; }
}

__global__ void k_minmax(const float* x, size_t n, unsigned* stats, int idx){
  __shared__ float smn[256], smx[256];
  size_t i = (size_t)blockIdx.x*blockDim.x + threadIdx.x;
  size_t stride = (size_t)gridDim.x*blockDim.x;
  float mn = 0.f, mx = 0.f;
  for (; i < n; i += stride){ float v = x[i]; mn = fminf(mn,v); mx = fmaxf(mx,v); }
  int t = threadIdx.x; smn[t] = mn; smx[t] = mx; __syncthreads();
  for (int off = 128; off; off >>= 1){
    if (t < off){ smn[t] = fminf(smn[t], smn[t+off]); smx[t] = fmaxf(smx[t], smx[t+off]); }
    __syncthreads();
  }
  if (t == 0){ atomicMin(&stats[2*idx], fenc(smn[0])); atomicMax(&stats[2*idx+1], fenc(smx[0])); }
}

__global__ void k_quant_act768(const float* x, int8_t* xq, int* xsum,
                               const unsigned* stats, int idx){
  __shared__ int ssum;
  int row = blockIdx.x, t = threadIdx.x;
  float s, zp; qparams(stats, idx, s, zp);
  if (t == 0) ssum = 0;
  __syncthreads();
  const float* xr = x + (size_t)row*CDIM;
  int8_t* qr = xq + (size_t)row*CDIM;
  int acc = 0;
  for (int j = 0; j < 3; j++){
    int c = t + 256*j;
    int q = (int)clampf(rintf(xr[c]/s)+zp, -128.f, 127.f);
    qr[c] = (int8_t)q; acc += q;
  }
  atomicAdd(&ssum, acc);
  __syncthreads();
  if (t == 0) xsum[row] = ssum;
}

__global__ void k_quant_qkv(const float* y, int8_t* yq, int* qsum, int* ksum,
                            const unsigned* stats){
  __shared__ int qs[NH], ks[NH];
  int row = blockIdx.x, t = threadIdx.x;
  float s, zp; qparams(stats, 1, s, zp);
  if (t < NH){ qs[t] = 0; ks[t] = 0; }
  __syncthreads();
  const float* yr = y + (size_t)row*C3;
  int8_t* qr = yq + (size_t)row*C3;
  for (int j = 0; j < 9; j++){
    int c = t + 256*j;
    int q = (int)clampf(rintf(yr[c]/s)+zp, -128.f, 127.f);
    qr[c] = (int8_t)q;
    int which = c / CDIM, h = (c % CDIM) / HD;
    if (which == 0)      atomicAdd(&qs[h], q);
    else if (which == 1) atomicAdd(&ks[h], q);
  }
  __syncthreads();
  if (t < NH){
    int b = row / SEQ, n = row % SEQ;
    size_t idx = (size_t)(b*NH + t)*SEQ + n;
    qsum[idx] = qs[t]; ksum[idx] = ks[t];
  }
}

__global__ void k_gemm_iu8(const int8_t* aq, const int* asum,
                           const int8_t* wq, const int* wsum, const float* wsc, const float* wzp,
                           const float* bias, float* out, int ldo,
                           unsigned* stats, int idxIn, int idxOut){
  __shared__ int8_t As[16*CDIM];
  int tid = threadIdx.x;
  int lane = tid & 31, wvid = tid >> 5;
  int m0 = blockIdx.y*16;
  int nb = blockIdx.x*256 + wvid*64;
  bool lo = lane < 16; int l15 = lane & 15;

  for (int i = 0; i < 6; i++){
    int c = tid + 128*i;
    async_b128(aq + (size_t)m0*CDIM + c*16, As + c*16);
  }
#if USE_ASYNC
  wait_async();
#endif
  __syncthreads();

  float sa, za; qparams(stats, idxIn, sa, za);
  v8i zi = {0,0,0,0,0,0,0,0};
  v8i acc[4] = {zi, zi, zi, zi};
  const int* Ai = (const int*)As;
  const int8_t* brow0 = wq + (size_t)(nb +  0 + l15)*CDIM;
  const int8_t* brow1 = wq + (size_t)(nb + 16 + l15)*CDIM;
  const int8_t* brow2 = wq + (size_t)(nb + 32 + l15)*CDIM;
  const int8_t* brow3 = wq + (size_t)(nb + 48 + l15)*CDIM;
  int abase = lo ? 0 : 8, bbase = lo ? 0 : 8;
  for (int k0 = 0; k0 < CDIM; k0 += 64){
    v8i av;
#pragma unroll
    for (int v = 0; v < 8; v++)
      av[v] = Ai[(l15*CDIM + k0 + 16*(v>>1) + 4*(v&1) + abase) >> 2];
    v8i b0, b1, b2, b3;
#pragma unroll
    for (int v = 0; v < 8; v++){
      int boff = k0 + 16*(v>>1) + 4*(v&1) + bbase;
      b0[v] = *(const int*)(brow0 + boff);
      b1[v] = *(const int*)(brow1 + boff);
      b2[v] = *(const int*)(brow2 + boff);
      b3[v] = *(const int*)(brow3 + boff);
    }
    acc[0] = __builtin_amdgcn_wmma_i32_16x16x64_iu8(true, av, true, b0, acc[0], false, false);
    acc[1] = __builtin_amdgcn_wmma_i32_16x16x64_iu8(true, av, true, b1, acc[1], false, false);
    acc[2] = __builtin_amdgcn_wmma_i32_16x16x64_iu8(true, av, true, b2, acc[2], false, false);
    acc[3] = __builtin_amdgcn_wmma_i32_16x16x64_iu8(true, av, true, b3, acc[3], false, false);
  }
  float asm8[8];
#pragma unroll
  for (int r = 0; r < 8; r++) asm8[r] = (float)asum[m0 + r + (lo ? 0 : 8)];
  float mn = 0.f, mx = 0.f;
#pragma unroll
  for (int nt = 0; nt < 4; nt++){
    int n = nb + nt*16 + l15;
    float swn = wsc[n], zwn = wzp[n], wsn = (float)wsum[n], bn = bias[n];
#pragma unroll
    for (int r = 0; r < 8; r++){
      int m = m0 + r + (lo ? 0 : 8);
      float S = (float)acc[nt][r];
      float val = sa*swn*(S - za*wsn - zwn*asm8[r] + 768.0f*za*zwn) + bn;
      out[(size_t)m*ldo + n] = val;
      mn = fminf(mn, val); mx = fmaxf(mx, val);
    }
  }
  for (int off = 16; off; off >>= 1){
    mn = fminf(mn, __shfl_xor(mn, off, 32));
    mx = fmaxf(mx, __shfl_xor(mx, off, 32));
  }
  if (lane == 0){ atomicMin(&stats[2*idxOut], fenc(mn)); atomicMax(&stats[2*idxOut+1], fenc(mx)); }
}

__device__ __forceinline__ v8i load_qA(const int8_t* yq, int b, int h, int n0, int lane){
  int l15 = lane & 15;
  int qi = n0 + l15; if (qi > SEQ-1) qi = SEQ-1;
  const int8_t* row = yq + (size_t)(b*SEQ + qi)*C3 + h*HD;
  int abase = (lane < 16) ? 0 : 8;
  v8i a;
#pragma unroll
  for (int v = 0; v < 8; v++) a[v] = *(const int*)(row + 16*(v>>1) + 4*(v&1) + abase);
  return a;
}
__device__ __forceinline__ v8i load_kB(const int8_t* yq, int b, int h, int m0, int lane){
  int l15 = lane & 15;
  int kj = m0 + l15; if (kj > SEQ-1) kj = SEQ-1;
  const int8_t* row = yq + (size_t)(b*SEQ + kj)*C3 + CDIM + h*HD;
  int bbase = (lane < 16) ? 0 : 8;
  v8i bb;
#pragma unroll
  for (int v = 0; v < 8; v++) bb[v] = *(const int*)(row + 16*(v>>1) + 4*(v&1) + bbase);
  return bb;
}

__global__ void k_attn_minmax(const int8_t* yq, const int* qsum, const int* ksum,
                              unsigned* stats){
  int lane = threadIdx.x, bh = blockIdx.x;
  int b = bh/NH, h = bh%NH, n0 = blockIdx.y*16;
  bool lo = lane < 16; int l15 = lane & 15;
  float s1, z1; qparams(stats, 1, s1, z1);
  float fac = s1*s1*0.125f, z64 = 64.f*z1*z1;
  v8i aF = load_qA(yq, b, h, n0, lane);
  float qs[8];
#pragma unroll
  for (int r = 0; r < 8; r++){
    int qi = n0 + r + (lo?0:8);
    qs[r] = (qi < SEQ) ? (float)qsum[(size_t)bh*SEQ + qi] : 0.f;
  }
  float mn = 0.f, mx = 0.f;
  for (int mt = 0; mt < NT; mt++){
    v8i bF = load_kB(yq, b, h, mt*16, lane);
    v8i c = {0,0,0,0,0,0,0,0};
    c = __builtin_amdgcn_wmma_i32_16x16x64_iu8(true, aF, true, bF, c, false, false);
    int kj = mt*16 + l15;
    float ksv = (kj < SEQ) ? (float)ksum[(size_t)bh*SEQ + kj] : 0.f;
#pragma unroll
    for (int r = 0; r < 8; r++){
      int qi = n0 + r + (lo?0:8);
      if (qi < SEQ && kj < SEQ){
        float raw = ((float)c[r] - z1*(ksv + qs[r]) + z64)*fac;
        mn = fminf(mn, raw); mx = fmaxf(mx, raw);
      }
    }
  }
  for (int off = 16; off; off >>= 1){
    mn = fminf(mn, __shfl_xor(mn, off, 32));
    mx = fmaxf(mx, __shfl_xor(mx, off, 32));
  }
  if (lane == 0){ atomicMin(&stats[4], fenc(mn)); atomicMax(&stats[5], fenc(mx)); }
}

__global__ void k_attn_rowstats(const int8_t* yq, const int* qsum, const int* ksum,
                                const unsigned* stats, float* rowmax, float* rowsum){
  int lane = threadIdx.x, bh = blockIdx.x;
  int b = bh/NH, h = bh%NH, n0 = blockIdx.y*16;
  bool lo = lane < 16; int l15 = lane & 15;
  float s1, z1; qparams(stats, 1, s1, z1);
  float sA, zA; qparams(stats, 2, sA, zA);
  float fac = s1*s1*0.125f, z64 = 64.f*z1*z1;
  v8i aF = load_qA(yq, b, h, n0, lane);
  float qs[8], rmx[8], rs[8];
#pragma unroll
  for (int r = 0; r < 8; r++){
    int qi = n0 + r + (lo?0:8);
    qs[r] = (qi < SEQ) ? (float)qsum[(size_t)bh*SEQ + qi] : 0.f;
    rmx[r] = -3.0e38f; rs[r] = 0.f;
  }
  for (int mt = 0; mt < NT; mt++){
    v8i bF = load_kB(yq, b, h, mt*16, lane);
    v8i c = {0,0,0,0,0,0,0,0};
    c = __builtin_amdgcn_wmma_i32_16x16x64_iu8(true, aF, true, bF, c, false, false);
    int kj = mt*16 + l15;
    float ksv = (kj < SEQ) ? (float)ksum[(size_t)bh*SEQ + kj] : 0.f;
#pragma unroll
    for (int r = 0; r < 8; r++){
      int qi = n0 + r + (lo?0:8);
      if (qi < SEQ && kj < SEQ){
        float raw = ((float)c[r] - z1*(ksv + qs[r]) + z64)*fac;
        float ai  = clampf(rintf(raw/sA)+zA, -128.f, 127.f) - zA;
        rmx[r] = fmaxf(rmx[r], ai);
      }
    }
  }
  for (int off = 1; off < 16; off <<= 1)
#pragma unroll
    for (int r = 0; r < 8; r++) rmx[r] = fmaxf(rmx[r], __shfl_xor(rmx[r], off, 32));
  float x0i = floorf(-0.6931f/sA);
  float bi  = floorf((0.96963238f/0.35815147f)/sA);
  float ci  = floorf((1.0f/0.35815147f)/(sA*sA));
  for (int mt = 0; mt < NT; mt++){
    v8i bF = load_kB(yq, b, h, mt*16, lane);
    v8i c = {0,0,0,0,0,0,0,0};
    c = __builtin_amdgcn_wmma_i32_16x16x64_iu8(true, aF, true, bF, c, false, false);
    int kj = mt*16 + l15;
    float ksv = (kj < SEQ) ? (float)ksum[(size_t)bh*SEQ + kj] : 0.f;
#pragma unroll
    for (int r = 0; r < 8; r++){
      int qi = n0 + r + (lo?0:8);
      if (qi < SEQ && kj < SEQ){
        float raw = ((float)c[r] - z1*(ksv + qs[r]) + z64)*fac;
        float ai  = clampf(rintf(raw/sA)+zA, -128.f, 127.f) - zA;
        float xi  = fmaxf(ai - rmx[r], 30.f*x0i);
        float qn  = floorf(xi/x0i);
        float rr  = xi - x0i*qn;
        float ei  = fmaxf(floorf((rr*(rr+bi)+ci)*exp2f(30.f-qn)), 0.f);
        rs[r] += ei;
      }
    }
  }
  for (int off = 1; off < 16; off <<= 1)
#pragma unroll
    for (int r = 0; r < 8; r++) rs[r] += __shfl_xor(rs[r], off, 32);
  if (l15 == 0){
    int base = lo ? 0 : 8;
    for (int r = 0; r < 8; r++){
      int qi = n0 + r + base;
      if (qi < SEQ){ rowmax[(size_t)bh*SEQ+qi] = rmx[r]; rowsum[(size_t)bh*SEQ+qi] = rs[r]; }
    }
  }
}

__global__ void k_attn_av(const int8_t* yq, const int* qsum, const int* ksum,
                          const float* rowmax, const float* rowsum,
                          unsigned* stats, float* o){
  __shared__ _Float16 wt[16][34];
  __shared__ int8_t vt[32*64];
  int lane = threadIdx.x, bh = blockIdx.x;
  int b = bh/NH, h = bh%NH, n0 = blockIdx.y*16;
  bool lo = lane < 16; int l15 = lane & 15;
  float s1, z1; qparams(stats, 1, s1, z1);
  float sA, zA; qparams(stats, 2, sA, zA);
  float fac = s1*s1*0.125f, z64 = 64.f*z1*z1;
  float x0i = floorf(-0.6931f/sA);
  float bi  = floorf((0.96963238f/0.35815147f)/sA);
  float ci  = floorf((1.0f/0.35815147f)/(sA*sA));
  v8i aF = load_qA(yq, b, h, n0, lane);
  float qs[8], rmx[8], rsm[8];
#pragma unroll
  for (int r = 0; r < 8; r++){
    int qi = n0 + r + (lo?0:8);
    bool v = qi < SEQ;
    qs[r]  = v ? (float)qsum[(size_t)bh*SEQ+qi] : 0.f;
    rmx[r] = v ? rowmax[(size_t)bh*SEQ+qi] : 0.f;
    rsm[r] = v ? rowsum[(size_t)bh*SEQ+qi] : 0.f;
  }
  v8f zf = {0.f,0.f,0.f,0.f,0.f,0.f,0.f,0.f};
  v8f acc[4] = {zf, zf, zf, zf};
  for (int ch = 0; ch < 19; ch++){
    __syncthreads();
    {
      int mg = ch*32 + lane; if (mg > SEQ-1) mg = SEQ-1;
      const int8_t* g = yq + (size_t)(b*SEQ + mg)*C3 + 2*CDIM + h*HD;
#pragma unroll
      for (int j = 0; j < 4; j++) async_b128(g + j*16, vt + lane*64 + j*16);
    }
    for (int sub = 0; sub < 2; sub++){
      int mt = ch*2 + sub;
      float w[8];
      if (mt < NT){
        v8i bF = load_kB(yq, b, h, mt*16, lane);
        v8i c = {0,0,0,0,0,0,0,0};
        c = __builtin_amdgcn_wmma_i32_16x16x64_iu8(true, aF, true, bF, c, false, false);
        int kj = mt*16 + l15;
        float ksv = (kj < SEQ) ? (float)ksum[(size_t)bh*SEQ+kj] : 0.f;
#pragma unroll
        for (int r = 0; r < 8; r++){
          int qi = n0 + r + (lo?0:8);
          float wv = 0.f;
          if (qi < SEQ && kj < SEQ){
            float raw = ((float)c[r] - z1*(ksv + qs[r]) + z64)*fac;
            float ai  = clampf(rintf(raw/sA)+zA, -128.f, 127.f) - zA;
            float xi  = fmaxf(ai - rmx[r], 30.f*x0i);
            float qn  = floorf(xi/x0i);
            float rr  = xi - x0i*qn;
            float ei  = fmaxf(floorf((rr*(rr+bi)+ci)*exp2f(30.f-qn)), 0.f);
            if (ei > 0.f && rsm[r] > 0.f){
              float sm  = rintf(rsm[r]/ei);
              float big = floorf(log2f(sm));
              if (sm - exp2f(big) >= exp2f(big - 1.f)) big += 1.f;
              if (big < 16.f) wv = exp2f(-clampf(big, 0.f, 15.f));
            }
          }
          w[r] = wv;
        }
      } else {
#pragma unroll
        for (int r = 0; r < 8; r++) w[r] = 0.f;
      }
#pragma unroll
      for (int r = 0; r < 8; r++) wt[r + (lo?0:8)][sub*16 + l15] = (_Float16)w[r];
    }
    __syncthreads();
#if USE_ASYNC
    wait_async();
#endif
    v16h aw;
#pragma unroll
    for (int v = 0; v < 8; v++){
      int kb = lo ? (v < 4 ? 2*v : 8 + 2*v) : (v < 4 ? 8 + 2*v : 16 + 2*v);
      aw[2*v]   = wt[l15][kb];
      aw[2*v+1] = wt[l15][kb+1];
    }
    for (int dt = 0; dt < 4; dt++){
      v16h bw;
#pragma unroll
      for (int v = 0; v < 8; v++){
        int kb = (v < 4 ? 2*v : 8 + 2*v) + (lo ? 0 : 8);
        for (int j = 0; j < 2; j++){
          int vq = (int)vt[(kb + j)*64 + dt*16 + l15];
          bw[2*v+j] = (_Float16)((float)vq - z1);
        }
      }
      acc[dt] = __builtin_amdgcn_wmma_f32_16x16x32_f16(false, aw, false, bw,
                                                       (short)0, acc[dt], false, false);
    }
  }
  float mn = 0.f, mx = 0.f;
#pragma unroll
  for (int r = 0; r < 8; r++){
    int qi = n0 + r + (lo?0:8);
    if (qi < SEQ){
      for (int dt = 0; dt < 4; dt++){
        float val = acc[dt][r]*s1;
        o[(size_t)(b*SEQ+qi)*CDIM + h*HD + dt*16 + l15] = val;
        mn = fminf(mn, val); mx = fmaxf(mx, val);
      }
    }
  }
  for (int off = 16; off; off >>= 1){
    mn = fminf(mn, __shfl_xor(mn, off, 32));
    mx = fmaxf(mx, __shfl_xor(mx, off, 32));
  }
  if (lane == 0){ atomicMin(&stats[6], fenc(mn)); atomicMax(&stats[7], fenc(mx)); }
}

__global__ void k_fq_inplace(float* x, size_t n, const unsigned* stats, int idx){
  float s, zp; qparams(stats, idx, s, zp);
  size_t i = (size_t)blockIdx.x*blockDim.x + threadIdx.x;
  size_t stride = (size_t)gridDim.x*blockDim.x;
  for (; i < n; i += stride){
    float q = clampf(rintf(x[i]/s)+zp, -128.f, 127.f);
    x[i] = (q - zp)*s;
  }
}

extern "C" void kernel_launch(void* const* d_in, const int* in_sizes, int n_in,
                              void* d_out, int out_size, void* d_ws, size_t ws_size,
                              hipStream_t stream){
  (void)in_sizes; (void)n_in; (void)out_size; (void)ws_size;
  const float* x      = (const float*)d_in[0];
  const float* w_qkv  = (const float*)d_in[1];
  const float* b_qkv  = (const float*)d_in[2];
  const float* w_proj = (const float*)d_in[3];
  const float* b_proj = (const float*)d_in[4];
  float* out = (float*)d_out;

  char* ws = (char*)d_ws;
  size_t cur = 0;
  auto alloc = [&](size_t bytes)->char*{
    char* p = ws + cur; cur = (cur + bytes + 255) & ~(size_t)255; return p;
  };
  unsigned* stats   = (unsigned*)alloc(16*4);
  int8_t* wq_qkv    = (int8_t*)alloc((size_t)C3*CDIM);
  int*    wsum_qkv  = (int*)alloc((size_t)C3*4);
  float*  wsc_qkv   = (float*)alloc((size_t)C3*4);
  float*  wzp_qkv   = (float*)alloc((size_t)C3*4);
  int8_t* wq_proj   = (int8_t*)alloc((size_t)CDIM*CDIM);
  int*    wsum_proj = (int*)alloc((size_t)CDIM*4);
  float*  wsc_proj  = (float*)alloc((size_t)CDIM*4);
  float*  wzp_proj  = (float*)alloc((size_t)CDIM*4);
  int8_t* xq        = (int8_t*)alloc((size_t)NTOK*CDIM);
  int*    xsum      = (int*)alloc((size_t)NTOK*4);
  float*  y         = (float*)alloc((size_t)NTOK*C3*4);
  int8_t* yq        = (int8_t*)alloc((size_t)NTOK*C3);
  int*    qsum      = (int*)alloc((size_t)NBH*SEQ*4);
  int*    ksum      = (int*)alloc((size_t)NBH*SEQ*4);
  float*  rowmax    = (float*)alloc((size_t)NBH*SEQ*4);
  float*  rowsum    = (float*)alloc((size_t)NBH*SEQ*4);
  float*  o  = y;
  int8_t* oq = xq;
  int*    osum = xsum;

  k_init_stats<<<1, 32, 0, stream>>>(stats);
  k_quant_w<<<C3 + CDIM, 256, 0, stream>>>(w_qkv, w_proj,
      wq_qkv, wsum_qkv, wsc_qkv, wzp_qkv, wq_proj, wsum_proj, wsc_proj, wzp_proj);
  k_minmax<<<2048, 256, 0, stream>>>(x, (size_t)NTOK*CDIM, stats, 0);
  k_quant_act768<<<NTOK, 256, 0, stream>>>(x, xq, xsum, stats, 0);
  k_gemm_iu8<<<dim3(C3/256, NTOK/16), 128, 0, stream>>>(xq, xsum,
      wq_qkv, wsum_qkv, wsc_qkv, wzp_qkv, b_qkv, y, C3, stats, 0, 1);
  k_quant_qkv<<<NTOK, 256, 0, stream>>>(y, yq, qsum, ksum, stats);
  k_attn_minmax<<<dim3(NBH, NT), 32, 0, stream>>>(yq, qsum, ksum, stats);
  k_attn_rowstats<<<dim3(NBH, NT), 32, 0, stream>>>(yq, qsum, ksum, stats, rowmax, rowsum);
  k_attn_av<<<dim3(NBH, NT), 32, 0, stream>>>(yq, qsum, ksum, rowmax, rowsum, stats, o);
  k_quant_act768<<<NTOK, 256, 0, stream>>>(o, oq, osum, stats, 3);
  k_gemm_iu8<<<dim3(CDIM/256, NTOK/16), 128, 0, stream>>>(oq, osum,
      wq_proj, wsum_proj, wsc_proj, wzp_proj, b_proj, out, CDIM, stats, 3, 4);
  k_fq_inplace<<<2048, 256, 0, stream>>>(out, (size_t)NTOK*CDIM, stats, 4);
}
